// MultiHeadAttention_6485400617325
// MI455X (gfx1250) — matrix operand order repaired and checked by reading, not compiled, not run on hardware
//
#include <hip/hip_runtime.h>
#include <hip/hip_bf16.h>

// ---------------------------------------------------------------------------
// MultiHeadAttention with relative position bias + causal mask + LogSoftmax
// B=2, S=2048, E=1024, H=16, D=64.  All GEMMs via v_wmma_f32_16x16x32_bf16.
// ---------------------------------------------------------------------------

#define Bc 2
#define Sc 2048
#define Ec 1024
#define Hc 16
#define Dc 64
#define BSE (Bc * Sc * Ec)   // 4,194,304
#define EE  (Ec * Ec)        // 1,048,576

typedef __attribute__((ext_vector_type(16))) __bf16 bf16x16;
typedef __attribute__((ext_vector_type(8)))  float  f32x8;

union Frag {
    bf16x16  v;
    unsigned u[8];
};

__device__ __forceinline__ unsigned short f2bf(float f) {
    union { float f; unsigned u; } x;
    x.f = f;
    unsigned r = x.u + 0x7FFFu + ((x.u >> 16) & 1u);  // round-to-nearest-even
    return (unsigned short)(r >> 16);
}

__device__ __forceinline__ unsigned pack_bf2(float lo, float hi) {
    return (unsigned)f2bf(lo) | ((unsigned)f2bf(hi) << 16);
}

// Load 8 bf16 at p0 and 8 bf16 at p1 (two b128 loads) into a fragment.
__device__ __forceinline__ void ld_frag(Frag& f, const unsigned short* p0,
                                        const unsigned short* p1) {
    uint4 a = *(const uint4*)p0;
    uint4 b = *(const uint4*)p1;
    f.u[0] = a.x; f.u[1] = a.y; f.u[2] = a.z; f.u[3] = a.w;
    f.u[4] = b.x; f.u[5] = b.y; f.u[6] = b.z; f.u[7] = b.w;
}

// ---------------------------------------------------------------------------
// fp32 -> bf16 conversion
// ---------------------------------------------------------------------------
__global__ void cvt_f32_bf16(const float* __restrict__ in,
                             unsigned short* __restrict__ out, int n) {
    int i = blockIdx.x * blockDim.x + threadIdx.x;
    if (i < n) out[i] = f2bf(in[i]);
}

// ---------------------------------------------------------------------------
// Y = X @ W^T + bias, written bf16 in split-head layout.
//   transposed == 0 : Y -> [B,H,S,D]   (used for Q, K)
//   transposed == 1 : Y -> [B,H,D,S]   (used for V; makes attn@V loads b128)
// X: [4096,1024] bf16 row-major.  W: [1024,1024] bf16 row-major (W[e*E+c]).
// One 16x32 output strip per wave (A fragment reused for two N tiles).
// ---------------------------------------------------------------------------
__global__ void __launch_bounds__(256)
gemm_proj(const unsigned short* __restrict__ X,
          const unsigned short* __restrict__ W,
          const float* __restrict__ bias,
          unsigned short* __restrict__ Y,
          int transposed) {
    int waveG  = blockIdx.x * (blockDim.x >> 5) + (threadIdx.x >> 5);
    int lane   = threadIdx.x & 31;
    int ln     = lane & 15;
    int hiHalf = lane >> 4;             // 0 or 1

    int tileN2 = waveG & 31;            // 32 strips of 32 columns
    int tileM  = waveG >> 5;            // 256 tiles of B*S

    int rowA  = tileM * 16 + ln;        // A row (striped over lanes)
    int colB0 = tileN2 * 32 + ln;       // first output feature e
    int colB1 = colB0 + 16;             // second output feature e

    const unsigned short* xrow  = X + (size_t)rowA  * Ec;
    const unsigned short* wrow0 = W + (size_t)colB0 * Ec;
    const unsigned short* wrow1 = W + (size_t)colB1 * Ec;

    f32x8 acc0 = {}, acc1 = {};
    for (int k0 = 0; k0 < Ec; k0 += 32) {
        Frag a, b0, b1;
        int abase = k0 + (hiHalf ? 8 : 0);      // A: 16x32 bf16 layout
        ld_frag(a, xrow + abase, xrow + abase + 16);
        int bbase = k0 + (hiHalf ? 8 : 0);      // B: 32x16 bf16 layout
        ld_frag(b0, wrow0 + bbase, wrow0 + bbase + 16);
        ld_frag(b1, wrow1 + bbase, wrow1 + bbase + 16);
        acc0 = __builtin_amdgcn_wmma_f32_16x16x32_bf16(
            false, a.v, false, b0.v, (short)0, acc0, false, false);
        acc1 = __builtin_amdgcn_wmma_f32_16x16x32_bf16(
            false, a.v, false, b1.v, (short)0, acc1, false, false);
    }

    for (int t = 0; t < 2; ++t) {
        int e = t ? colB1 : colB0;
        const f32x8& acc = t ? acc1 : acc0;
        float bval = bias[e];
        int h = e >> 6;                 // e / D
        int d = e & 63;                 // e % D
        for (int i = 0; i < 8; ++i) {
            int r = tileM * 16 + i + 8 * hiHalf;    // C: M = i + 8*(lane/16)
            int bb = r >> 11;                       // r / S
            int s  = r & (Sc - 1);                  // r % S
            unsigned short yv = f2bf(acc[i] + bval);
            if (transposed)
                Y[(((size_t)(bb * Hc + h) * Dc + d) * Sc) + s] = yv;
            else
                Y[(((size_t)(bb * Hc + h) * Sc + s) * Dc) + d] = yv;
        }
    }
}

// ---------------------------------------------------------------------------
// Attention: per workgroup = one (b,h) and one 16-row query block. 4 waves.
// Dynamic LDS holds the full 16x2048 fp32 logit block (128 KB).
//   Phase 1: logits = QK^T/sqrt(D) + pos_bias, causal mask (-1e9)
//   Phase 2: per-row c = max + log(sum exp)   (LogSoftmax constant)
//   Phase 3: out = (logits - c) @ V           (faithful: log-probs times V)
//            waves split the 2048-key range; partials reduced through LDS.
// ---------------------------------------------------------------------------
__global__ void __launch_bounds__(128)
attn_kernel(const unsigned short* __restrict__ Q,   // [B,H,S,D] bf16
            const unsigned short* __restrict__ K,   // [B,H,S,D] bf16
            const unsigned short* __restrict__ Vt,  // [B,H,D,S] bf16
            const float* __restrict__ posb,         // [H, 2S-1]
            float* __restrict__ out) {              // [B,S,E]
    extern __shared__ float logitsS[];              // 16 * 2048 floats
    __shared__ float red[128];
    __shared__ float rowred[16];
    __shared__ float cvals[16];
    __shared__ float pOut[4][16][Dc];               // per-wave partial outputs

    int qt = blockIdx.x & 127;          // S/16 query tiles
    int bh = blockIdx.x >> 7;           // b*H + h
    int h  = bh & 15;
    int b  = bh >> 4;
    int q0 = qt * 16;

    int tid    = threadIdx.x;
    int w      = tid >> 5;
    int lane   = tid & 31;
    int ln     = lane & 15;
    int hiHalf = lane >> 4;

    const unsigned short* Qb  = Q  + (size_t)bh * Sc * Dc;
    const unsigned short* Kb  = K  + (size_t)bh * Sc * Dc;
    const unsigned short* Vtb = Vt + (size_t)bh * Dc * Sc;
    const float* pb = posb + (size_t)h * (2 * Sc - 1);

    // --- Q fragments for the two 32-wide D chunks (A: 16x32 layout) ---
    Frag aq0, aq1;
    {
        const unsigned short* qp = Qb + (size_t)(q0 + ln) * Dc;
        int b0 = (hiHalf ? 8 : 0);
        ld_frag(aq0, qp + b0,      qp + b0 + 16);
        ld_frag(aq1, qp + 32 + b0, qp + 32 + b0 + 16);
    }

    // --- Phase 1: logits; each wave covers 32 of 128 key tiles ---
    for (int kt = w * 32; kt < w * 32 + 32; ++kt) {
        int key = kt * 16 + ln;                       // B column = key index
        const unsigned short* kp = Kb + (size_t)key * Dc;
        Frag b0, b1;
        int off = hiHalf ? 8 : 0;                     // B: 32x16 layout
        ld_frag(b0, kp + off,      kp + off + 16);
        ld_frag(b1, kp + 32 + off, kp + 32 + off + 16);
        f32x8 acc = {};
        acc = __builtin_amdgcn_wmma_f32_16x16x32_bf16(
            false, aq0.v, false, b0.v, (short)0, acc, false, false);
        acc = __builtin_amdgcn_wmma_f32_16x16x32_bf16(
            false, aq1.v, false, b1.v, (short)0, acc, false, false);
        for (int i = 0; i < 8; ++i) {
            int m = i + 8 * hiHalf;
            int q = q0 + m;
            float val = acc[i] * 0.125f + pb[q - key + (Sc - 1)];
            if (key > q) val = -1e9f;                 // causal masked_fill
            logitsS[m * Sc + key] = val;
        }
    }
    __syncthreads();

    // --- Phase 2: per-row max and log-sum-exp ---
    {
        int row = tid & 15;
        int grp = tid >> 4;                            // 8 column groups
        const float* lr = logitsS + row * Sc + grp * 256;
        float mx = -3.4e38f;
        for (int c = 0; c < 256; ++c) mx = fmaxf(mx, lr[c]);
        red[grp * 16 + row] = mx;
        __syncthreads();
        if (tid < 16) {
            float m2 = red[tid];
            for (int g = 1; g < 8; ++g) m2 = fmaxf(m2, red[g * 16 + tid]);
            rowred[tid] = m2;
        }
        __syncthreads();
        float rmax = rowred[row];
        float sum = 0.f;
        for (int c = 0; c < 256; ++c) sum += __expf(lr[c] - rmax);
        red[grp * 16 + row] = sum;
        __syncthreads();
        if (tid < 16) {
            float s2 = 0.f;
            for (int g = 0; g < 8; ++g) s2 += red[g * 16 + tid];
            cvals[tid] = rowred[tid] + __logf(s2);
        }
        __syncthreads();
    }

    // --- Phase 3: out = (logits - c) @ V ---
    // Wave w owns keys [w*512, (w+1)*512): builds each bf16 A-fragment once,
    // then 4 WMMAs (one per 16-wide D tile) against contiguous V^T fragments.
    {
        float cm = cvals[ln];                          // A row = ln for all lanes
        f32x8 acc[4] = {{}, {}, {}, {}};
        for (int ks = w * 16; ks < w * 16 + 16; ++ks) {
            int k0 = ks * 32;
            Frag a;
            const float* lrow = logitsS + ln * Sc + k0;
            int koff = hiHalf ? 8 : 0;                 // A: 16x32 layout
            #pragma unroll
            for (int j = 0; j < 4; ++j) {
                a.u[j]     = pack_bf2(lrow[koff + 2 * j]      - cm,
                                      lrow[koff + 2 * j + 1]  - cm);
                a.u[4 + j] = pack_bf2(lrow[16 + koff + 2 * j]     - cm,
                                      lrow[16 + koff + 2 * j + 1] - cm);
            }
            int kbase = k0 + (hiHalf ? 8 : 0);         // B: 32x16 layout
            #pragma unroll
            for (int dt = 0; dt < 4; ++dt) {
                int d = dt * 16 + ln;                  // B column = head dim
                const unsigned short* vrow = Vtb + (size_t)d * Sc;
                Frag bv;
                ld_frag(bv, vrow + kbase, vrow + kbase + 16);
                acc[dt] = __builtin_amdgcn_wmma_f32_16x16x32_bf16(
                    false, a.v, false, bv.v, (short)0, acc[dt], false, false);
            }
        }
        #pragma unroll
        for (int dt = 0; dt < 4; ++dt)
            for (int i = 0; i < 8; ++i)
                pOut[w][i + 8 * hiHalf][dt * 16 + ln] = acc[dt][i];
    }
    __syncthreads();

    // --- Cross-wave reduction of the 16x64 output tile; write fp32 out ---
    for (int idx = tid * 8; idx < tid * 8 + 8; ++idx) {   // 1024 elems / 128 thr
        int m = idx >> 6;
        int d = idx & 63;
        float s4 = pOut[0][m][d] + pOut[1][m][d] + pOut[2][m][d] + pOut[3][m][d];
        int s = q0 + m;
        out[(size_t)(b * Sc + s) * Ec + (h * Dc + d)] = s4;
    }
}

// ---------------------------------------------------------------------------
// Launch
// ---------------------------------------------------------------------------
extern "C" void kernel_launch(void* const* d_in, const int* in_sizes, int n_in,
                              void* d_out, int out_size, void* d_ws, size_t ws_size,
                              hipStream_t stream) {
    const float* query = (const float*)d_in[0];
    const float* key   = (const float*)d_in[1];
    const float* value = (const float*)d_in[2];
    const float* Wq    = (const float*)d_in[3];
    const float* bq    = (const float*)d_in[4];
    const float* Wk    = (const float*)d_in[5];
    const float* bk    = (const float*)d_in[6];
    const float* Wv    = (const float*)d_in[7];
    const float* bv    = (const float*)d_in[8];
    const float* posb  = (const float*)d_in[9];
    // d_in[10..12] (idx0, idx1, mask) are recomputed analytically on device.

    float* out = (float*)d_out;

    // Workspace layout (bf16 / ushort), total ~54.5 MB
    unsigned short* Xq  = (unsigned short*)d_ws;
    unsigned short* Xk  = Xq  + BSE;
    unsigned short* Xv  = Xk  + BSE;
    unsigned short* Wqb = Xv  + BSE;
    unsigned short* Wkb = Wqb + EE;
    unsigned short* Wvb = Wkb + EE;
    unsigned short* Qs  = Wvb + EE;
    unsigned short* Ks  = Qs  + BSE;
    unsigned short* Vts = Ks  + BSE;

    // 1) fp32 -> bf16 conversions
    cvt_f32_bf16<<<BSE / 256, 256, 0, stream>>>(query, Xq, BSE);
    cvt_f32_bf16<<<BSE / 256, 256, 0, stream>>>(key,   Xk, BSE);
    cvt_f32_bf16<<<BSE / 256, 256, 0, stream>>>(value, Xv, BSE);
    cvt_f32_bf16<<<EE  / 256, 256, 0, stream>>>(Wq,    Wqb, EE);
    cvt_f32_bf16<<<EE  / 256, 256, 0, stream>>>(Wk,    Wkb, EE);
    cvt_f32_bf16<<<EE  / 256, 256, 0, stream>>>(Wv,    Wvb, EE);

    // 2) Projections: (256 M-tiles * 32 N-strips) / 8 waves = 1024 blocks
    gemm_proj<<<1024, 256, 0, stream>>>(Xq, Wqb, bq, Qs,  0);
    gemm_proj<<<1024, 256, 0, stream>>>(Xk, Wkb, bk, Ks,  0);
    gemm_proj<<<1024, 256, 0, stream>>>(Xv, Wvb, bv, Vts, 1);   // V transposed

    // 3) Attention: B*H*(S/16) = 4096 blocks, 128 threads, 128 KB dynamic LDS
    const int smem = 16 * Sc * (int)sizeof(float);
    hipFuncSetAttribute((const void*)attn_kernel,
                        hipFuncAttributeMaxDynamicSharedMemorySize, smem);
    attn_kernel<<<Bc * Hc * (Sc / 16), 128, smem, stream>>>(Qs, Ks, Vts, posb, out);
}
